// BNN_58961311039944
// MI455X (gfx1250) — 16-bit and 8-bit matrix operand order repaired and checked by reading, compiled, not run on hardware
//
#include <hip/hip_runtime.h>
#include <stdint.h>

typedef _Float16 f16_t;
typedef f16_t v8h   __attribute__((ext_vector_type(8)));
typedef f16_t v16h  __attribute__((ext_vector_type(16)));
typedef float  v8f  __attribute__((ext_vector_type(8)));
typedef int    v2i  __attribute__((ext_vector_type(2)));
typedef int    v4i  __attribute__((ext_vector_type(4)));
typedef int    v8i  __attribute__((ext_vector_type(8)));

static constexpr size_t OFF_W0H  = 0;
static constexpr size_t OFF_W1   = OFF_W0H  + (size_t)4096*4096*2;
static constexpr size_t OFF_W2   = OFF_W1   + (size_t)4096*4096;
static constexpr size_t OFF_W3   = OFF_W2   + (size_t)4096*4096;
static constexpr size_t OFF_H    = OFF_W3   + (size_t)1024*4096;
static constexpr size_t OFF_ACTA = OFF_H    + (size_t)4096*4096*4;
static constexpr size_t OFF_ACTB = OFF_ACTA + (size_t)4096*4096;
static constexpr size_t OFF_PART = OFF_ACTB + (size_t)4096*4096;
static constexpr size_t OFF_SS   = OFF_PART + (size_t)2*32*4096*4;

__global__ __launch_bounds__(256) void sign_f16_k(const float* __restrict__ W,
                                                  f16_t* __restrict__ o, int n) {
  int i = blockIdx.x * 256 + threadIdx.x;
  if (i < n) {
    float x = W[i];
    o[i] = x > 0.f ? (f16_t)1.0f : (x < 0.f ? (f16_t)(-1.0f) : (f16_t)0.0f);
  }
}

__global__ __launch_bounds__(256) void sign_i8_k(const float* __restrict__ W,
                                                 int8_t* __restrict__ o, int n) {
  int i = blockIdx.x * 256 + threadIdx.x;
  if (i < n) {
    float x = W[i];
    o[i] = x > 0.f ? (int8_t)1 : (x < 0.f ? (int8_t)-1 : (int8_t)0);
  }
}

__global__ __launch_bounds__(256) void gemm_l1_f16x2(const float* __restrict__ X,
                                                     const f16_t* __restrict__ Wh,
                                                     float* __restrict__ H,
                                                     int K, int N) {
  __shared__ f16_t Bs[128 * 32];

  const int lane = threadIdx.x & 31;
  const int wave = threadIdx.x >> 5;
  const int lm = lane & 15, lh = lane >> 4;
  const int m0 = blockIdx.y * 128 + wave * 16;
  const int nblk = blockIdx.x * 128;

  const int cn = threadIdx.x >> 1;
  const int ch = threadIdx.x & 1;
  const f16_t* wrow = Wh + (size_t)(nblk + cn) * K + ch * 16;
  f16_t* bdst = &Bs[cn * 32 + ch * 16];

  v8f acc[8] = {};

  for (int k0 = 0; k0 < K; k0 += 32) {
    *(v16h*)bdst = *(const v16h*)(wrow + k0);
    __syncthreads();

    const float* pA = X + (size_t)(m0 + lm) * K + (k0 + lh * 16);
    float4 f0 = *(const float4*)(pA + 0);
    float4 f1 = *(const float4*)(pA + 4);
    float4 f2 = *(const float4*)(pA + 8);
    float4 f3 = *(const float4*)(pA + 12);
    float xs[16] = {f0.x, f0.y, f0.z, f0.w, f1.x, f1.y, f1.z, f1.w,
                    f2.x, f2.y, f2.z, f2.w, f3.x, f3.y, f3.z, f3.w};
    v16h ahi, alo;
#pragma unroll
    for (int j = 0; j < 16; ++j) {
      float x = xs[j];
      f16_t h1 = (f16_t)x;
      float r = x - (float)h1;
      ahi[j] = h1;
      alo[j] = (f16_t)r;
    }

    v16h bfr[8];
#pragma unroll
    for (int ni = 0; ni < 8; ++ni)
      bfr[ni] = *(const v16h*)&Bs[(ni * 16 + lm) * 32 + lh * 16];

#pragma unroll
    for (int ni = 0; ni < 8; ++ni) {
      acc[ni] = __builtin_amdgcn_wmma_f32_16x16x32_f16(
          false, ahi, false, bfr[ni], (short)0, acc[ni], false, false);
      acc[ni] = __builtin_amdgcn_wmma_f32_16x16x32_f16(
          false, alo, false, bfr[ni], (short)0, acc[ni], false, false);
    }
    __syncthreads();
  }

#pragma unroll
  for (int ni = 0; ni < 8; ++ni)
#pragma unroll
    for (int r = 0; r < 8; ++r)
      H[(size_t)(m0 + r + 8 * lh) * N + (nblk + ni * 16 + lm)] = acc[ni][r];
}

__global__ __launch_bounds__(256) void gemm_i8(const int8_t* __restrict__ A,
                                               const int8_t* __restrict__ W,
                                               int* __restrict__ H,
                                               int K, int N) {
  const int lane = threadIdx.x & 31;
  const int wave = threadIdx.x >> 5;
  const int lm = lane & 15, lh = lane >> 4;
  const int m0 = blockIdx.y * 128 + (wave >> 1) * 32;
  const int n0 = blockIdx.x * 128 + (wave & 1) * 64;

  v8i acc[2][4] = {};

  for (int k0 = 0; k0 < K; k0 += 64) {
    v8i af[2];
#pragma unroll
    for (int mi = 0; mi < 2; ++mi) {
      const int8_t* pA = A + (size_t)(m0 + mi * 16 + lm) * K + (k0 + lh * 16);
      v2i q0 = *(const v2i*)(pA + 0);
      v2i q1 = *(const v2i*)(pA + 8);
      v2i q2 = *(const v2i*)(pA + 32);
      v2i q3 = *(const v2i*)(pA + 40);
      v8i a = {q0[0], q0[1], q1[0], q1[1], q2[0], q2[1], q3[0], q3[1]};
      af[mi] = a;
    }
    v8i bfr[4];
#pragma unroll
    for (int ni = 0; ni < 4; ++ni) {
      const int8_t* pB = W + (size_t)(n0 + ni * 16 + lm) * K + (k0 + lh * 16);
      v4i r0 = *(const v4i*)(pB);
      v4i r1 = *(const v4i*)(pB + 32);
      v8i b = {r0[0], r0[1], r0[2], r0[3], r1[0], r1[1], r1[2], r1[3]};
      bfr[ni] = b;
    }
#pragma unroll
    for (int mi = 0; mi < 2; ++mi)
#pragma unroll
      for (int ni = 0; ni < 4; ++ni)
        acc[mi][ni] = __builtin_amdgcn_wmma_i32_16x16x64_iu8(
            true, af[mi], true, bfr[ni], acc[mi][ni], false, false);
  }
#pragma unroll
  for (int mi = 0; mi < 2; ++mi)
#pragma unroll
    for (int ni = 0; ni < 4; ++ni)
#pragma unroll
      for (int r = 0; r < 8; ++r)
        H[(size_t)(m0 + mi * 16 + r + 8 * lh) * N + (n0 + ni * 16 + lm)] = acc[mi][ni][r];
}

__global__ __launch_bounds__(256) void colstats_part(const void* __restrict__ Hv, int isInt,
                                                     int N, int rows, int nchunk,
                                                     float* __restrict__ part) {
  int col = blockIdx.x * 256 + threadIdx.x;
  int c = blockIdx.y;
  size_t base = (size_t)c * rows * N + col;
  float s = 0.f, s2 = 0.f;
  if (isInt) {
    const int* H = (const int*)Hv;
    for (int r = 0; r < rows; ++r) { float v = (float)H[base + (size_t)r * N]; s += v; s2 += v * v; }
  } else {
    const float* H = (const float*)Hv;
    for (int r = 0; r < rows; ++r) { float v = H[base + (size_t)r * N]; s += v; s2 += v * v; }
  }
  part[(size_t)c * N + col] = s;
  part[(size_t)(nchunk + c) * N + col] = s2;
}

__global__ __launch_bounds__(256) void colstats_final(const float* __restrict__ part,
                                                      const float* __restrict__ g,
                                                      const float* __restrict__ b,
                                                      float* __restrict__ ss,
                                                      int N, int nchunk, float invB) {
  int col = blockIdx.x * 256 + threadIdx.x;
  float s = 0.f, s2 = 0.f;
  for (int c = 0; c < nchunk; ++c) {
    s  += part[(size_t)c * N + col];
    s2 += part[(size_t)(nchunk + c) * N + col];
  }
  float mu  = s * invB;
  float var = s2 * invB - mu * mu;
  float sc  = g[col] * rsqrtf(var + 1e-5f);
  ss[col]     = sc;
  ss[N + col] = b[col] - mu * sc;
}

__global__ __launch_bounds__(256) void bn_sign_k(const void* __restrict__ Hv, int isInt,
                                                 const float* __restrict__ ss,
                                                 int8_t* __restrict__ act, int N) {
  size_t i = (size_t)blockIdx.x * 256 + threadIdx.x;
  int col = (int)(i % (size_t)N);
  float h = isInt ? (float)((const int*)Hv)[i] : ((const float*)Hv)[i];
  float v = h * ss[col] + ss[N + col];
  act[i] = v > 0.f ? (int8_t)1 : (v < 0.f ? (int8_t)-1 : (int8_t)0);
}

__global__ __launch_bounds__(256) void bn_out_k(const int* __restrict__ H,
                                                const float* __restrict__ ss,
                                                float* __restrict__ out, int N) {
  size_t i = (size_t)blockIdx.x * 256 + threadIdx.x;
  int col = (int)(i % (size_t)N);
  out[i] = (float)H[i] * ss[col] + ss[N + col];
}

extern "C" void kernel_launch(void* const* d_in, const int* in_sizes, int n_in,
                              void* d_out, int out_size, void* d_ws, size_t ws_size,
                              hipStream_t stream) {
  (void)in_sizes; (void)n_in; (void)out_size; (void)ws_size;
  const float* x  = (const float*)d_in[0];
  const float* W0 = (const float*)d_in[1];
  const float* g0 = (const float*)d_in[2];
  const float* b0 = (const float*)d_in[3];
  const float* W1 = (const float*)d_in[4];
  const float* g1 = (const float*)d_in[5];
  const float* b1 = (const float*)d_in[6];
  const float* W2 = (const float*)d_in[7];
  const float* g2 = (const float*)d_in[8];
  const float* b2 = (const float*)d_in[9];
  const float* W3 = (const float*)d_in[10];
  const float* g3 = (const float*)d_in[11];
  const float* b3 = (const float*)d_in[12];

  char* ws = (char*)d_ws;
  f16_t*  W0h  = (f16_t*)(ws + OFF_W0H);
  int8_t* W1s  = (int8_t*)(ws + OFF_W1);
  int8_t* W2s  = (int8_t*)(ws + OFF_W2);
  int8_t* W3s  = (int8_t*)(ws + OFF_W3);
  float*  Hf   = (float*)(ws + OFF_H);
  int*    Hi   = (int*)(ws + OFF_H);
  int8_t* actA = (int8_t*)(ws + OFF_ACTA);
  int8_t* actB = (int8_t*)(ws + OFF_ACTB);
  float*  part = (float*)(ws + OFF_PART);
  float*  ss   = (float*)(ws + OFF_SS);

  const int B = 4096, K = 4096;
  const int NCH = 32, ROWS = B / NCH;
  const float invB = 1.0f / (float)B;

  sign_f16_k<<<65536, 256, 0, stream>>>(W0, W0h, 4096 * 4096);
  sign_i8_k <<<65536, 256, 0, stream>>>(W1, W1s, 4096 * 4096);
  sign_i8_k <<<65536, 256, 0, stream>>>(W2, W2s, 4096 * 4096);
  sign_i8_k <<<16384, 256, 0, stream>>>(W3, W3s, 1024 * 4096);

  gemm_l1_f16x2 <<<dim3(32, 32), 256, 0, stream>>>(x, W0h, Hf, K, 4096);
  colstats_part <<<dim3(16, NCH), 256, 0, stream>>>(Hf, 0, 4096, ROWS, NCH, part);
  colstats_final<<<16, 256, 0, stream>>>(part, g0, b0, ss, 4096, NCH, invB);
  bn_sign_k     <<<65536, 256, 0, stream>>>(Hf, 0, ss, actA, 4096);

  gemm_i8       <<<dim3(32, 32), 256, 0, stream>>>(actA, W1s, Hi, K, 4096);
  colstats_part <<<dim3(16, NCH), 256, 0, stream>>>(Hi, 1, 4096, ROWS, NCH, part);
  colstats_final<<<16, 256, 0, stream>>>(part, g1, b1, ss, 4096, NCH, invB);
  bn_sign_k     <<<65536, 256, 0, stream>>>(Hi, 1, ss, actB, 4096);

  gemm_i8       <<<dim3(32, 32), 256, 0, stream>>>(actB, W2s, Hi, K, 4096);
  colstats_part <<<dim3(16, NCH), 256, 0, stream>>>(Hi, 1, 4096, ROWS, NCH, part);
  colstats_final<<<16, 256, 0, stream>>>(part, g2, b2, ss, 4096, NCH, invB);
  bn_sign_k     <<<65536, 256, 0, stream>>>(Hi, 1, ss, actA, 4096);

  gemm_i8       <<<dim3(8, 32), 256, 0, stream>>>(actA, W3s, Hi, K, 1024);
  colstats_part <<<dim3(4, NCH), 256, 0, stream>>>(Hi, 1, 1024, ROWS, NCH, part);
  colstats_final<<<4, 256, 0, stream>>>(part, g3, b3, ss, 1024, NCH, invB);
  bn_out_k      <<<16384, 256, 0, stream>>>(Hi, ss, (float*)d_out, 1024);
}
